// BERT_MultiHeadAttention_70093866270729
// MI455X (gfx1250) — matrix operand order repaired and checked by reading, not compiled, not run on hardware
//
#include <hip/hip_runtime.h>

// ---------------------------------------------------------------------------
// BERT MHA forward for gfx1250 (MI455X): bf16 WMMA everywhere, flash-attention
// style streaming softmax (no materialized S x S scores).
// Shapes: B=8, S=1024, D=768, H=12, Dh=64, 3D=2304.
// ---------------------------------------------------------------------------

typedef __attribute__((ext_vector_type(16))) __bf16 v16bf;
typedef __attribute__((ext_vector_type(8)))  float  v8f;

#define B_   8
#define S_   1024
#define D_   768
#define H_   12
#define DH_  64
#define D3_  2304
#define NTOK 8192            // B_*S_

__device__ __forceinline__ unsigned short f2bf(float f) {
    union { float f; unsigned u; } c; c.f = f;
    unsigned u = c.u;
    unsigned r = u + 0x7FFFu + ((u >> 16) & 1u);   // round-to-nearest-even
    return (unsigned short)(r >> 16);
}

// A-fragment (16xK=32, bf16): lane holds row M=lane%16.
// VGPR j<4: K = 8*half + 2j,+1 ; VGPR j>=4: K = 16 + 8*half + 2(j-4),+1
__device__ __forceinline__ v16bf load_fragA(const unsigned short* base, int ld,
                                            int row, int k0) {
    union { v16bf v; unsigned u[8]; } f;
    const int hl = (threadIdx.x >> 4) & 1;
    const unsigned short* p = base + (size_t)row * ld + k0 + 8 * hl;
#pragma unroll
    for (int j = 0; j < 4; ++j) f.u[j]     = *(const unsigned*)(p + 2 * j);
#pragma unroll
    for (int j = 0; j < 4; ++j) f.u[4 + j] = *(const unsigned*)(p + 16 + 2 * j);
    return f.v;
}

// B-fragment (K=32 x 16, bf16): lane holds col N=lane%16.
// VGPR j<4: K = 8*half + 2j,+1 ; VGPR j>=4: K = 16 + 8*half + 2(j-4),+1
__device__ __forceinline__ v16bf load_fragB(const unsigned short* base, int ld,
                                            int col, int k0) {
    union { v16bf v; unsigned u[8]; } f;
    const int hl = (threadIdx.x >> 4) & 1;
    const unsigned short* p = base + (size_t)col * ld + k0 + 8 * hl;
#pragma unroll
    for (int j = 0; j < 4; ++j) f.u[j]     = *(const unsigned*)(p + 2 * j);
#pragma unroll
    for (int j = 0; j < 4; ++j) f.u[4 + j] = *(const unsigned*)(p + 16 + 2 * j);
    return f.v;
}

__device__ __forceinline__ v8f wmma_bf16(v16bf a, v16bf b, v8f c) {
    return __builtin_amdgcn_wmma_f32_16x16x32_bf16(false, a, false, b,
                                                   (short)0, c, false, false);
}

__device__ __forceinline__ float redmax16(float v) {
#pragma unroll
    for (int m = 8; m >= 1; m >>= 1) v = fmaxf(v, __shfl_xor(v, m, 32));
    return v;
}
__device__ __forceinline__ float redsum16(float v) {
#pragma unroll
    for (int m = 8; m >= 1; m >>= 1) v += __shfl_xor(v, m, 32);
    return v;
}

// ---------------------------------------------------------------------------
// Software-pipelined 32x64 GEMM tile: acc[2][4] += A[32xK] * Bt[64 rows x K].
// Next k-step's fragments are loaded before this step's WMMAs so loads stay
// in flight across the 8-WMMA burst (avoids s_wait_loadcnt 0 on every step).
// ---------------------------------------------------------------------------
__device__ __forceinline__ void gemm_tile_32x64(const unsigned short* __restrict__ A,
                                                const unsigned short* __restrict__ Bt,
                                                int row0, int col0, int K,
                                                v8f acc[2][4]) {
    const int lm = threadIdx.x & 15;
    v16bf a0 = load_fragA(A, K, row0 + lm, 0);
    v16bf a1 = load_fragA(A, K, row0 + 16 + lm, 0);
    v16bf b0 = load_fragB(Bt, K, col0 + lm, 0);
    v16bf b1 = load_fragB(Bt, K, col0 + 16 + lm, 0);
    v16bf b2 = load_fragB(Bt, K, col0 + 32 + lm, 0);
    v16bf b3 = load_fragB(Bt, K, col0 + 48 + lm, 0);
    for (int k0 = 32; k0 <= K; k0 += 32) {
        const int kn = (k0 < K) ? k0 : 0;   // last iter: dummy reload, no branch
        v16bf na0 = load_fragA(A, K, row0 + lm, kn);
        v16bf na1 = load_fragA(A, K, row0 + 16 + lm, kn);
        v16bf nb0 = load_fragB(Bt, K, col0 + lm, kn);
        v16bf nb1 = load_fragB(Bt, K, col0 + 16 + lm, kn);
        v16bf nb2 = load_fragB(Bt, K, col0 + 32 + lm, kn);
        v16bf nb3 = load_fragB(Bt, K, col0 + 48 + lm, kn);
        acc[0][0] = wmma_bf16(a0, b0, acc[0][0]);
        acc[0][1] = wmma_bf16(a0, b1, acc[0][1]);
        acc[0][2] = wmma_bf16(a0, b2, acc[0][2]);
        acc[0][3] = wmma_bf16(a0, b3, acc[0][3]);
        acc[1][0] = wmma_bf16(a1, b0, acc[1][0]);
        acc[1][1] = wmma_bf16(a1, b1, acc[1][1]);
        acc[1][2] = wmma_bf16(a1, b2, acc[1][2]);
        acc[1][3] = wmma_bf16(a1, b3, acc[1][3]);
        a0 = na0; a1 = na1; b0 = nb0; b1 = nb1; b2 = nb2; b3 = nb3;
    }
}

// ---------------------------------------------------------------------------
// Conversion kernels
// ---------------------------------------------------------------------------
__global__ __launch_bounds__(256) void k_f32_to_bf16(const float* __restrict__ src,
                                                     unsigned short* __restrict__ dst,
                                                     int n) {
    int i = blockIdx.x * blockDim.x + threadIdx.x;
    if (i < n) dst[i] = f2bf(src[i]);
}

// W is (K x N) fp32 row-major; Wt is (N x K) bf16 row-major.
__global__ __launch_bounds__(256) void k_transpose_bf16(const float* __restrict__ W,
                                                        unsigned short* __restrict__ Wt,
                                                        int K, int N) {
    int i = blockIdx.x * blockDim.x + threadIdx.x;
    if (i >= K * N) return;
    int n = i / K, k = i % K;
    Wt[i] = f2bf(W[(size_t)k * N + n]);
}

// ---------------------------------------------------------------------------
// QKV projection: C[8192,2304] = Xb @ Wqkv + b ; scatter into Q,K [B,H,S,Dh]
// and Vt [B,H,Dh,S] (bf16). One wave -> 32x64 tile.  9216 waves.
// ---------------------------------------------------------------------------
__global__ __launch_bounds__(256) void k_gemm_qkv(const unsigned short* __restrict__ Xb,
                                                  const unsigned short* __restrict__ Wt,
                                                  const float* __restrict__ bias,
                                                  unsigned short* __restrict__ Qb,
                                                  unsigned short* __restrict__ Kb,
                                                  unsigned short* __restrict__ Vt) {
    const int lane = threadIdx.x & 31, lm = lane & 15, hl = lane >> 4;
    const int wave = (blockIdx.x * blockDim.x + threadIdx.x) >> 5;
    const int mtile = wave / 36;       // 256 tiles of 32 rows
    const int ng    = wave % 36;       // 36 groups of 64 cols
    v8f acc[2][4] = {};
    gemm_tile_32x64(Xb, Wt, mtile * 32, ng * 64, D_, acc);
#pragma unroll
    for (int rs = 0; rs < 2; ++rs)
#pragma unroll
        for (int j = 0; j < 4; ++j) {
            const int col = ng * 64 + j * 16 + lm;
            const float bv = bias[col];
#pragma unroll
            for (int r = 0; r < 8; ++r) {
                const int tok = mtile * 32 + rs * 16 + r + 8 * hl;
                const int bb = tok >> 10, ss = tok & 1023;
                const unsigned short v16 = f2bf(acc[rs][j][r] + bv);
                if (col < D_) {
                    int hh = col >> 6, dh = col & 63;
                    Qb[(((size_t)(bb * H_ + hh)) * S_ + ss) * DH_ + dh] = v16;
                } else if (col < 2 * D_) {
                    int c = col - D_; int hh = c >> 6, dh = c & 63;
                    Kb[(((size_t)(bb * H_ + hh)) * S_ + ss) * DH_ + dh] = v16;
                } else {
                    int c = col - 2 * D_; int hh = c >> 6, dh = c & 63;
                    Vt[(((size_t)(bb * H_ + hh)) * DH_ + dh) * S_ + ss] = v16;
                }
            }
        }
}

// ---------------------------------------------------------------------------
// Flash attention: one wave per (b,h,16-query tile). 6144 waves.
// ---------------------------------------------------------------------------
__global__ __launch_bounds__(256) void k_flash(const unsigned short* __restrict__ Qb,
                                               const unsigned short* __restrict__ Kb,
                                               const unsigned short* __restrict__ Vt,
                                               const int* __restrict__ mask,
                                               unsigned short* __restrict__ Ctx) {
    __shared__ unsigned short Pb[8][16 * 32];       // per-wave P scratch (1 KB)
    const int lane = threadIdx.x & 31, lm = lane & 15, hl = lane >> 4;
    const int wib  = threadIdx.x >> 5;
    const int wave = blockIdx.x * 8 + wib;
    const int qtile = wave & 63;                    // S_/16 = 64
    const int bh    = wave >> 6;                    // 0..95
    const int b = bh / H_, h = bh % H_;
    const unsigned short* Qh = Qb + (size_t)bh * (S_ * DH_);
    const unsigned short* Kh = Kb + (size_t)bh * (S_ * DH_);
    const unsigned short* Vh = Vt + (size_t)bh * (DH_ * S_);
    const int* mk = mask + b * S_;
    unsigned short* P = &Pb[wib][0];
    const int q0 = qtile * 16;

    const v16bf qa0 = load_fragA(Qh, DH_, q0 + lm, 0);
    const v16bf qa1 = load_fragA(Qh, DH_, q0 + lm, 32);

    float Mr[8], Lr[8];
#pragma unroll
    for (int r = 0; r < 8; ++r) { Mr[r] = -1e30f; Lr[r] = 0.0f; }
    v8f o[4] = {};

    for (int kb = 0; kb < S_; kb += 32) {
        // ---- scores: two 16x16 tiles, K = Dh = 64 (two wmma steps each) ----
        v8f s0 = {}, s1 = {};
        {
            v16bf kf0 = load_fragB(Kh, DH_, kb + lm, 0);
            v16bf kf1 = load_fragB(Kh, DH_, kb + lm, 32);
            s0 = wmma_bf16(qa0, kf0, s0);
            s0 = wmma_bf16(qa1, kf1, s0);
            v16bf kf2 = load_fragB(Kh, DH_, kb + 16 + lm, 0);
            v16bf kf3 = load_fragB(Kh, DH_, kb + 16 + lm, 32);
            s1 = wmma_bf16(qa0, kf2, s1);
            s1 = wmma_bf16(qa1, kf3, s1);
        }
        // ---- issue V-fragment loads now; they overlap the softmax VALU work
        v16bf vf0 = load_fragB(Vh, S_, 0 * 16 + lm, kb);
        v16bf vf1 = load_fragB(Vh, S_, 1 * 16 + lm, kb);
        v16bf vf2 = load_fragB(Vh, S_, 2 * 16 + lm, kb);
        v16bf vf3 = load_fragB(Vh, S_, 3 * 16 + lm, kb);

        const bool msk0 = (mk[kb + lm] == 0);
        const bool msk1 = (mk[kb + 16 + lm] == 0);

        // ---- online softmax per row (row r+8*hl lives in one 16-lane half) --
        float e0[8], e1[8], rs[8];
#pragma unroll
        for (int r = 0; r < 8; ++r) {
            float a0 = msk0 ? -1000.0f : s0[r] * 0.125f;
            float a1 = msk1 ? -1000.0f : s1[r] * 0.125f;
            float mc = redmax16(fmaxf(a0, a1));
            float mn = fmaxf(Mr[r], mc);
            float sc = __expf(Mr[r] - mn);
            float x0 = __expf(a0 - mn);
            float x1 = __expf(a1 - mn);
            Lr[r] = Lr[r] * sc + redsum16(x0 + x1);
            Mr[r] = mn;
            rs[r] = sc; e0[r] = x0; e1[r] = x1;
        }
#pragma unroll
        for (int r = 0; r < 8; ++r) {
            o[0][r] *= rs[r]; o[1][r] *= rs[r];
            o[2][r] *= rs[r]; o[3][r] *= rs[r];
            const int prow = r + 8 * hl;
            P[prow * 32 + lm]      = f2bf(e0[r]);
            P[prow * 32 + 16 + lm] = f2bf(e1[r]);
        }
        asm volatile("s_wait_dscnt 0" ::: "memory");   // wave-local LDS RAW

        // ---- P (16x32) @ V (32x64): 4 wmma ----
        const v16bf pf = load_fragA(P, 32, lm, 0);
        o[0] = wmma_bf16(pf, vf0, o[0]);
        o[1] = wmma_bf16(pf, vf1, o[1]);
        o[2] = wmma_bf16(pf, vf2, o[2]);
        o[3] = wmma_bf16(pf, vf3, o[3]);
    }

#pragma unroll
    for (int r = 0; r < 8; ++r) {
        const float inv = 1.0f / Lr[r];
        o[0][r] *= inv; o[1][r] *= inv; o[2][r] *= inv; o[3][r] *= inv;
    }
#pragma unroll
    for (int t = 0; t < 4; ++t)
#pragma unroll
        for (int r = 0; r < 8; ++r) {
            const int row = q0 + r + 8 * hl;
            const int col = h * DH_ + t * 16 + lm;
            Ctx[((size_t)(b * S_ + row)) * D_ + col] = f2bf(o[t][r]);
        }
}

// ---------------------------------------------------------------------------
// Output projection: out[8192,768] = Ctx @ Wout + b (fp32 out). 3072 waves.
// ---------------------------------------------------------------------------
__global__ __launch_bounds__(256) void k_gemm_out(const unsigned short* __restrict__ Cb,
                                                  const unsigned short* __restrict__ Wt,
                                                  const float* __restrict__ bias,
                                                  float* __restrict__ out) {
    const int lane = threadIdx.x & 31, lm = lane & 15, hl = lane >> 4;
    const int wave = (blockIdx.x * blockDim.x + threadIdx.x) >> 5;
    const int mtile = wave / 12;       // 256 tiles of 32 rows
    const int ng    = wave % 12;       // 12 groups of 64 cols
    v8f acc[2][4] = {};
    gemm_tile_32x64(Cb, Wt, mtile * 32, ng * 64, D_, acc);
#pragma unroll
    for (int rs = 0; rs < 2; ++rs)
#pragma unroll
        for (int j = 0; j < 4; ++j) {
            const int col = ng * 64 + j * 16 + lm;
            const float bv = bias[col];
#pragma unroll
            for (int r = 0; r < 8; ++r) {
                const int tok = mtile * 32 + rs * 16 + r + 8 * hl;
                out[(size_t)tok * D_ + col] = acc[rs][j][r] + bv;
            }
        }
}

// ---------------------------------------------------------------------------
extern "C" void kernel_launch(void* const* d_in, const int* in_sizes, int n_in,
                              void* d_out, int out_size, void* d_ws, size_t ws_size,
                              hipStream_t stream) {
    const float* x    = (const float*)d_in[0];   // [8,1024,768]
    const int*   mask = (const int*)  d_in[1];   // [8,1,1,1024]
    const float* Wqkv = (const float*)d_in[2];   // [768,2304]
    const float* bqkv = (const float*)d_in[3];   // [2304]
    const float* Wout = (const float*)d_in[4];   // [768,768]
    const float* bout = (const float*)d_in[5];   // [768]
    float* out = (float*)d_out;

    char* ws = (char*)d_ws;
    size_t off = 0;
    auto alloc = [&](size_t bytes) -> unsigned short* {
        unsigned short* p = (unsigned short*)(ws + off);
        off = (off + bytes + 255) & ~(size_t)255;
        return p;
    };
    unsigned short* Xb    = alloc((size_t)NTOK * D_ * 2);        // x bf16
    unsigned short* WqkvT = alloc((size_t)D3_ * D_ * 2);         // [2304,768]
    unsigned short* WoutT = alloc((size_t)D_ * D_ * 2);          // [768,768]
    unsigned short* Qb    = alloc((size_t)B_ * H_ * S_ * DH_ * 2);  // [B,H,S,Dh]
    unsigned short* Kb    = alloc((size_t)B_ * H_ * S_ * DH_ * 2);  // [B,H,S,Dh]
    unsigned short* Vt    = alloc((size_t)B_ * H_ * DH_ * S_ * 2);  // [B,H,Dh,S]
    unsigned short* Ctxb  = alloc((size_t)NTOK * D_ * 2);        // [B,S,D]

    const int NX = NTOK * D_;                     // 6291456
    k_f32_to_bf16<<<(NX + 255) / 256, 256, 0, stream>>>(x, Xb, NX);
    k_transpose_bf16<<<(D_ * D3_ + 255) / 256, 256, 0, stream>>>(Wqkv, WqkvT, D_, D3_);
    k_transpose_bf16<<<(D_ * D_ + 255) / 256, 256, 0, stream>>>(Wout, WoutT, D_, D_);

    // 256 mtiles * 36 ngroups = 9216 waves -> 1152 blocks of 256
    k_gemm_qkv<<<1152, 256, 0, stream>>>(Xb, WqkvT, bqkv, Qb, Kb, Vt);

    // 96 (b,h) * 64 qtiles = 6144 waves -> 768 blocks of 256
    k_flash<<<768, 256, 0, stream>>>(Qb, Kb, Vt, mask, Ctxb);

    // 256 mtiles * 12 ngroups = 3072 waves -> 384 blocks of 256
    k_gemm_out<<<384, 256, 0, stream>>>(Ctxb, WoutT, bout, out);
}
